// RBF_MultiHeadSelfAttention_81905026335128
// MI455X (gfx1250) — compile-verified, matrix operand order checked by reading, not run on hardware
//
#include <hip/hip_runtime.h>
#include <hip/hip_bf16.h>

// ---------------- constants ----------------
constexpr int BB = 2;
constexpr int SS = 2048;
constexpr int EE = 1024;
constexpr int HH = 16;
constexpr int DD = 64;
constexpr float SCORE_SCALE = 8.0f;  // sqrt(64)
constexpr int LDS_STRIDE = 2052;     // 2048 + 4 pad floats -> conflict-free WMMA A reads

typedef __bf16 bf16_t;
typedef __attribute__((ext_vector_type(16))) __bf16 v16bf;
typedef __attribute__((ext_vector_type(8)))  __bf16 v8bf;
typedef __attribute__((ext_vector_type(4)))  __bf16 v4bf;
typedef __attribute__((ext_vector_type(8)))  float  v8f;
typedef __attribute__((ext_vector_type(4)))  float  v4f;

union BFFrag { v16bf v; v8bf h[2]; };

// ---------------- WMMA helpers ----------------
// A/B fragment (16x32, bf16, row-major src [rows x ld], K contiguous):
// lane<16 : row=lane,    elems 0..7 = K 0..7,  elems 8..15 = K 16..23
// lane>=16: row=lane-16, elems 0..7 = K 8..15, elems 8..15 = K 24..31
__device__ inline v16bf load_frag_bf16(const bf16_t* base, int ld) {
  int lane = threadIdx.x & 31;
  int row  = lane & 15;
  int kb   = (lane < 16) ? 0 : 8;
  const bf16_t* p = base + (long long)row * ld + kb;
  BFFrag f;
  f.h[0] = *(const v8bf*)(p);
  f.h[1] = *(const v8bf*)(p + 16);
  return f.v;
}

// Same fragment but source is f32 (global or LDS); convert to bf16 in regs.
__device__ inline v16bf load_frag_f32(const float* base, int ld) {
  int lane = threadIdx.x & 31;
  int row  = lane & 15;
  int kb   = (lane < 16) ? 0 : 8;
  const float* p = base + (long long)row * ld + kb;
  BFFrag f;
#pragma unroll
  for (int i = 0; i < 8; ++i) {
    f.h[0][i] = (__bf16)p[i];
    f.h[1][i] = (__bf16)p[16 + i];
  }
  return f.v;
}

__device__ inline v8f wmma_bf16(v16bf a, v16bf b, v8f c) {
  return __builtin_amdgcn_wmma_f32_16x16x32_bf16(false, a, false, b, (short)0, c,
                                                 false, false);
}

#define V8F_ZERO {0.f,0.f,0.f,0.f,0.f,0.f,0.f,0.f}

// Stage a 16x16 f32 C-tile into per-wave LDS (256 floats) in row-major order.
// C/D layout: VGPR r holds rows r (lanes 0-15) and r+8 (lanes 16-31), col = lane&15.
__device__ inline void stage_tile(float* stg, const v8f& acc, float addend) {
  int lane = threadIdx.x & 31;
  int n = lane & 15, mb = (lane < 16) ? 0 : 8;
#pragma unroll
  for (int r = 0; r < 8; ++r) stg[(mb + r) * 16 + n] = acc[r] + addend;
}

// Emit the staged tile as bf16 with coalesced 16B row stores (2 lanes per row).
// Wave-internal LDS RAW is safe: DS ops from one wave are in order.
__device__ inline void store_tile_bf16(const float* stg, bf16_t* dst, long long ld) {
  int lane = threadIdx.x & 31;
  int row  = lane >> 1;
  int half = (lane & 1) * 8;
  const float* src = stg + row * 16 + half;
  v8bf o;
#pragma unroll
  for (int i = 0; i < 8; ++i) o[i] = (__bf16)src[i];
  *(v8bf*)(dst + (long long)row * ld + half) = o;
}

// ---------------- elementwise helpers ----------------
__global__ void cvt_bf16_kernel(const float* __restrict__ in, bf16_t* __restrict__ out,
                                int n4) {
  for (int i = blockIdx.x * blockDim.x + threadIdx.x; i < n4; i += gridDim.x * blockDim.x) {
    v4f v = *((const v4f*)in + i);
    v4bf o = {(__bf16)v[0], (__bf16)v[1], (__bf16)v[2], (__bf16)v[3]};
    *((v4bf*)out + i) = o;
  }
}

// Mt[e*64+d] = M[d*64+e]  (transpose so GEMM B operand is [N=e, K=d] row-major)
__global__ void mt_kernel(const float* __restrict__ M, bf16_t* __restrict__ Mt) {
  for (int idx = threadIdx.x; idx < DD * DD; idx += blockDim.x) {
    int d = idx >> 6, e = idx & 63;
    Mt[e * 64 + d] = (__bf16)M[idx];
  }
}

// Vt[b,h,d,s] = V[b,s,h*64+d]
__global__ void vt_kernel(const bf16_t* __restrict__ V, bf16_t* __restrict__ Vt) {
  long long idx = (long long)blockIdx.x * blockDim.x + threadIdx.x;
  if (idx >= (long long)BB * SS * EE) return;
  int e = (int)(idx & (EE - 1));
  int s = (int)((idx >> 10) & (SS - 1));
  int b = (int)(idx >> 21);
  int h = e >> 6, d = e & 63;
  Vt[(((long long)(b * HH + h) * DD + d) << 11) + s] = V[idx];
}

// diag[b,h,s] = sum_e (sum_d X[d] * M[d,e]) * X[e]   (X = per-head row of Q or K)
__global__ __launch_bounds__(256) void diag_kernel(const bf16_t* __restrict__ X,
                                                   const float* __restrict__ Mf,
                                                   float* __restrict__ out) {
  __shared__ float Ms[DD * DD];
  for (int i = threadIdx.x; i < DD * DD; i += blockDim.x) Ms[i] = Mf[i];
  __syncthreads();
  int tid = blockIdx.x * blockDim.x + threadIdx.x;  // ordered (b,h,s)
  int s = tid & (SS - 1);
  int h = (tid >> 11) & (HH - 1);
  int b = tid >> 15;
  const bf16_t* row = X + ((long long)(b * SS + s) << 10) + h * DD;
  float r[DD];
#pragma unroll
  for (int d = 0; d < DD; ++d) r[d] = (float)row[d];
  float acc = 0.f;
  for (int e = 0; e < DD; ++e) {
    float m = 0.f;
#pragma unroll
    for (int d = 0; d < DD; ++d) m += r[d] * Ms[d * DD + e];
    acc += m * r[e];
  }
  out[tid] = acc;  // layout [B,H,S]
}

// ---------------- GEMM: Y(bf16)[4096,1024] = A(f32) @ W(bf16,[N,K])^T + bias ----------------
__global__ __launch_bounds__(256) void proj_gemm_kernel(const float* __restrict__ A,
                                                        const bf16_t* __restrict__ Bw,
                                                        const float* __restrict__ bias,
                                                        bf16_t* __restrict__ C) {
  __shared__ float stg[8 * 256];
  const int N = EE, K = EE;
  int w  = threadIdx.x >> 5;
  int m0 = (blockIdx.y * 2 + (w >> 2)) * 16;
  int n0 = (blockIdx.x * 4 + (w & 3)) * 16;
  v8f acc = V8F_ZERO;
  for (int k = 0; k < K; k += 32) {
    __builtin_prefetch(A + (long long)m0 * K + k + 64, 0, 1);
    v16bf a = load_frag_f32(A + (long long)m0 * K + k, K);
    v16bf b = load_frag_bf16(Bw + (long long)n0 * K + k, K);
    acc = wmma_bf16(a, b, acc);
  }
  float bv = bias[n0 + (threadIdx.x & 15)];
  stage_tile(stg + w * 256, acc, bv);
  store_tile_bf16(stg + w * 256, C + (long long)m0 * N + n0, N);
}

// ---------------- qM = Q @ M per head (K=64) ----------------
__global__ __launch_bounds__(256) void qm_gemm_kernel(const bf16_t* __restrict__ Q,
                                                      const bf16_t* __restrict__ Mt,
                                                      bf16_t* __restrict__ qM) {
  __shared__ float stg[8 * 256];
  int h  = blockIdx.x;
  int w  = threadIdx.x >> 5;
  int m0 = (blockIdx.y * 2 + (w >> 2)) * 16;
  int ct = w & 3;
  v8f acc = V8F_ZERO;
#pragma unroll
  for (int k = 0; k < DD; k += 32) {
    v16bf a = load_frag_bf16(Q + (long long)m0 * EE + h * DD + k, EE);
    v16bf b = load_frag_bf16(Mt + (ct * 16) * DD + k, DD);
    acc = wmma_bf16(a, b, acc);
  }
  stage_tile(stg + w * 256, acc, 0.f);
  store_tile_bf16(stg + w * 256, qM + (long long)m0 * EE + h * DD + ct * 16, EE);
}

// ---------------- fused scores -> softmax -> attn write -> attn@V ----------------
// grid: (S/32, B*H), block 256 (8 waves). 32 q-rows per workgroup.
__global__ __launch_bounds__(256) void attn_kernel(const bf16_t* __restrict__ qM,
                                                   const bf16_t* __restrict__ Kb,
                                                   const bf16_t* __restrict__ Vt,
                                                   const float* __restrict__ qMq,
                                                   const float* __restrict__ kMk,
                                                   float* __restrict__ attn,
                                                   bf16_t* __restrict__ ctx) {
  __shared__ float sc[32 * LDS_STRIDE];  // ~256.5 KB scores tile
  __shared__ float stg[8 * 256];         // per-wave store staging
  __shared__ float inv_sum[32];
  int bh = blockIdx.y;
  int b  = bh >> 4;
  int h  = bh & 15;
  int q0 = blockIdx.x * 32;
  int w    = threadIdx.x >> 5;
  int lane = threadIdx.x & 31;

  // -------- phase 1: scores tile [32 x 2048] via WMMA (qM @ K^T), K-dim = 64 --------
  {
    int rt = w & 1;
    int m0 = q0 + rt * 16;
    const bf16_t* arow = qM + ((long long)(b * SS) + m0) * EE + h * DD;
    v16bf a0 = load_frag_bf16(arow + 0, EE);
    v16bf a1 = load_frag_bf16(arow + 32, EE);
    int mb = (lane < 16) ? 0 : 8;
    float qv[8];
#pragma unroll
    for (int r = 0; r < 8; ++r) qv[r] = qMq[(long long)bh * SS + m0 + mb + r];
    for (int ct = (w >> 1); ct < SS / 16; ct += 4) {
      int n0 = ct * 16;
      const bf16_t* brow = Kb + ((long long)(b * SS) + n0) * EE + h * DD;
      v16bf b0 = load_frag_bf16(brow + 0, EE);
      v16bf b1 = load_frag_bf16(brow + 32, EE);
      v8f acc = V8F_ZERO;
      acc = wmma_bf16(a0, b0, acc);
      acc = wmma_bf16(a1, b1, acc);
      int n = n0 + (lane & 15);
      float kk = kMk[(long long)bh * SS + n];
#pragma unroll
      for (int r = 0; r < 8; ++r)
        sc[(rt * 16 + mb + r) * LDS_STRIDE + n] = -SCORE_SCALE * (qv[r] + kk - 2.0f * acc[r]);
    }
  }
  __syncthreads();

  // -------- phase 2: row softmax (each wave owns 4 rows; float4 LDS sweeps) --------
  for (int rr = 0; rr < 4; ++rr) {
    int row = w * 4 + rr;
    v4f* p4 = (v4f*)(sc + row * LDS_STRIDE);
    v4f mx4 = {-1e30f, -1e30f, -1e30f, -1e30f};
    for (int c = lane; c < SS / 4; c += 32) {
      v4f v = p4[c];
#pragma unroll
      for (int j = 0; j < 4; ++j) mx4[j] = fmaxf(mx4[j], v[j]);
    }
    float mx = fmaxf(fmaxf(mx4[0], mx4[1]), fmaxf(mx4[2], mx4[3]));
#pragma unroll
    for (int o = 16; o; o >>= 1) mx = fmaxf(mx, __shfl_xor(mx, o, 32));
    float sum = 0.f;
    for (int c = lane; c < SS / 4; c += 32) {
      v4f v = p4[c];
      v4f e;
#pragma unroll
      for (int j = 0; j < 4; ++j) e[j] = __expf(v[j] - mx);
      p4[c] = e;
      sum += e[0] + e[1] + e[2] + e[3];
    }
#pragma unroll
    for (int o = 16; o; o >>= 1) sum += __shfl_xor(sum, o, 32);
    if (lane == 0) inv_sum[row] = 1.0f / sum;
  }
  __syncthreads();

  // -------- phase 2b: normalize in LDS + stream attn tile to HBM as b128 --------
  {
    float* gout = attn + ((long long)bh * SS + q0) * SS;
    for (int f = threadIdx.x; f < 32 * (SS / 4); f += 256) {
      int row = f >> 9;             // 512 float4 per row
      int c4  = (f & 511) * 4;
      float inv = inv_sum[row];
      v4f* p = (v4f*)(sc + row * LDS_STRIDE + c4);
      v4f v = *p;
      v4f o = {v[0] * inv, v[1] * inv, v[2] * inv, v[3] * inv};
      *p = o;
      *(v4f*)(gout + (long long)row * SS + c4) = o;
    }
  }
  __syncthreads();

  // -------- phase 3: ctx tile [32 x 64] = attn @ V (K = 2048), one 16x16 tile/wave --------
  {
    int rt = w >> 2;
    int ct = w & 3;
    v8f acc = V8F_ZERO;
    const bf16_t* vbase = Vt + ((long long)bh * DD + ct * 16) * SS;
    const float*  abase = sc + (rt * 16) * LDS_STRIDE;
    for (int k = 0; k < SS; k += 32) {
      v16bf a  = load_frag_f32(abase + k, LDS_STRIDE);  // LDS f32 -> bf16
      v16bf bf = load_frag_bf16(vbase + k, SS);
      acc = wmma_bf16(a, bf, acc);
    }
    stage_tile(stg + w * 256, acc, 0.f);
    store_tile_bf16(stg + w * 256,
                    ctx + ((long long)(b * SS) + q0 + rt * 16) * EE + h * DD + ct * 16,
                    EE);
  }
}

// ---------------- out = ctx @ Wo^T + bo + x (f32, coalesced float4 stores) ----------------
__global__ __launch_bounds__(256) void out_gemm_kernel(const bf16_t* __restrict__ A,
                                                       const bf16_t* __restrict__ Bw,
                                                       const float* __restrict__ bias,
                                                       const float* __restrict__ resid,
                                                       float* __restrict__ C) {
  __shared__ float stg[8 * 256];
  const int N = EE, K = EE;
  int w  = threadIdx.x >> 5;
  int m0 = (blockIdx.y * 2 + (w >> 2)) * 16;
  int n0 = (blockIdx.x * 4 + (w & 3)) * 16;
  v8f acc = V8F_ZERO;
  for (int k = 0; k < K; k += 32) {
    v16bf a = load_frag_bf16(A + (long long)m0 * K + k, K);
    v16bf b = load_frag_bf16(Bw + (long long)n0 * K + k, K);
    acc = wmma_bf16(a, b, acc);
  }
  float* ws = stg + w * 256;
  stage_tile(ws, acc, 0.f);
  int lane = threadIdx.x & 31;
#pragma unroll
  for (int i = 0; i < 2; ++i) {
    int f = lane + i * 32;         // 64 float4 pieces: 16 rows x 4 quads
    int row = f >> 2;
    int q   = (f & 3) * 4;
    const float* src = ws + row * 16 + q;
    long long o = (long long)(m0 + row) * N + n0 + q;
    v4f acc4 = *(const v4f*)src;
    v4f bb   = *(const v4f*)(bias + n0 + q);
    v4f rr   = *(const v4f*)(resid + o);
    v4f out4 = {acc4[0] + bb[0] + rr[0], acc4[1] + bb[1] + rr[1],
                acc4[2] + bb[2] + rr[2], acc4[3] + bb[3] + rr[3]};
    *(v4f*)(C + o) = out4;
  }
}

// ---------------- LayerNorm in place on d_out rows (float4 path) ----------------
__global__ __launch_bounds__(256) void ln_kernel(float* __restrict__ out,
                                                 const float* __restrict__ gamma,
                                                 const float* __restrict__ beta) {
  __shared__ float red[16];
  long long row = blockIdx.x;
  v4f* p4 = (v4f*)(out + row * EE);
  int w = threadIdx.x >> 5, lane = threadIdx.x & 31;
  v4f v = p4[threadIdx.x];
  float s  = v[0] + v[1] + v[2] + v[3];
  float s2 = v[0] * v[0] + v[1] * v[1] + v[2] * v[2] + v[3] * v[3];
#pragma unroll
  for (int o = 16; o; o >>= 1) {
    s  += __shfl_xor(s, o, 32);
    s2 += __shfl_xor(s2, o, 32);
  }
  if (lane == 0) { red[w] = s; red[8 + w] = s2; }
  __syncthreads();
  if (threadIdx.x == 0) {
    float ts = 0.f, ts2 = 0.f;
    for (int i = 0; i < 8; ++i) { ts += red[i]; ts2 += red[8 + i]; }
    red[0] = ts; red[8] = ts2;
  }
  __syncthreads();
  float mu  = red[0] * (1.0f / EE);
  float var = red[8] * (1.0f / EE) - mu * mu;
  float inv = rsqrtf(var + 1e-5f);
  v4f g = ((const v4f*)gamma)[threadIdx.x];
  v4f be = ((const v4f*)beta)[threadIdx.x];
  v4f o4 = {(v[0] - mu) * inv * g[0] + be[0], (v[1] - mu) * inv * g[1] + be[1],
            (v[2] - mu) * inv * g[2] + be[2], (v[3] - mu) * inv * g[3] + be[3]};
  p4[threadIdx.x] = o4;
}

// ---------------- host launcher ----------------
extern "C" void kernel_launch(void* const* d_in, const int* in_sizes, int n_in,
                              void* d_out, int out_size, void* d_ws, size_t ws_size,
                              hipStream_t stream) {
  (void)in_sizes; (void)n_in; (void)out_size; (void)ws_size;
  const float* x     = (const float*)d_in[0];
  const float* Wq    = (const float*)d_in[1];
  const float* bq    = (const float*)d_in[2];
  const float* Wk    = (const float*)d_in[3];
  const float* bk    = (const float*)d_in[4];
  const float* Wv    = (const float*)d_in[5];
  const float* bv    = (const float*)d_in[6];
  const float* Mf    = (const float*)d_in[7];
  const float* Wo    = (const float*)d_in[8];
  const float* bo    = (const float*)d_in[9];
  const float* gamma = (const float*)d_in[10];
  const float* beta  = (const float*)d_in[11];

  float* outp  = (float*)d_out;                // [B,S,E]
  float* attnp = outp + (size_t)BB * SS * EE;  // [B,H,S,S]

  char* ws = (char*)d_ws;
  size_t off = 0;
  auto take = [&](size_t bytes) -> char* {
    char* p = ws + off;
    off += (bytes + 255) & ~(size_t)255;
    return p;
  };
  const size_t NW = (size_t)EE * EE;       // 1M weights
  const size_t NA = (size_t)BB * SS * EE;  // 4M activations
  bf16_t* Wqb = (bf16_t*)take(NW * 2);
  bf16_t* Wkb = (bf16_t*)take(NW * 2);
  bf16_t* Wvb = (bf16_t*)take(NW * 2);
  bf16_t* Wob = (bf16_t*)take(NW * 2);
  bf16_t* Mtb = (bf16_t*)take((size_t)DD * DD * 2);
  bf16_t* Qb  = (bf16_t*)take(NA * 2);
  bf16_t* Kb  = (bf16_t*)take(NA * 2);
  bf16_t* Vb  = (bf16_t*)take(NA * 2);
  bf16_t* qMb = (bf16_t*)take(NA * 2);
  bf16_t* Vtb = (bf16_t*)take(NA * 2);
  bf16_t* ctx = (bf16_t*)take(NA * 2);
  float*  qMq = (float*)take((size_t)BB * HH * SS * 4);
  float*  kMk = (float*)take((size_t)BB * HH * SS * 4);

  // 1) weights f32 -> bf16 (float4 path), M transpose
  cvt_bf16_kernel<<<512, 256, 0, stream>>>(Wq, Wqb, (int)(NW / 4));
  cvt_bf16_kernel<<<512, 256, 0, stream>>>(Wk, Wkb, (int)(NW / 4));
  cvt_bf16_kernel<<<512, 256, 0, stream>>>(Wv, Wvb, (int)(NW / 4));
  cvt_bf16_kernel<<<512, 256, 0, stream>>>(Wo, Wob, (int)(NW / 4));
  mt_kernel<<<1, 256, 0, stream>>>(Mf, Mtb);

  // 2) QKV projections (WMMA)
  dim3 gp(EE / 64, (BB * SS) / 32);
  proj_gemm_kernel<<<gp, 256, 0, stream>>>(x, Wqb, bq, Qb);
  proj_gemm_kernel<<<gp, 256, 0, stream>>>(x, Wkb, bk, Kb);
  proj_gemm_kernel<<<gp, 256, 0, stream>>>(x, Wvb, bv, Vb);

  // 3) qM = Q @ M per head (WMMA), diagonals qMq/kMk, V transpose
  qm_gemm_kernel<<<dim3(HH, (BB * SS) / 32), 256, 0, stream>>>(Qb, Mtb, qMb);
  diag_kernel<<<(BB * HH * SS) / 256, 256, 0, stream>>>(Qb, Mf, qMq);
  diag_kernel<<<(BB * HH * SS) / 256, 256, 0, stream>>>(Kb, Mf, kMk);
  vt_kernel<<<(int)(NA / 256), 256, 0, stream>>>(Vb, Vtb);

  // 4) fused scores -> softmax -> attn -> attn@V (WMMA both GEMMs)
  attn_kernel<<<dim3(SS / 32, BB * HH), 256, 0, stream>>>(qMb, Kb, Vtb, qMq, kMk,
                                                          attnp, ctx);

  // 5) output projection + bias + residual, then LayerNorm in place
  out_gemm_kernel<<<gp, 256, 0, stream>>>(ctx, Wob, bo, x, outp);
  ln_kernel<<<BB * SS, 256, 0, stream>>>(outp, gamma, beta);
}
